// MultiheadAttention_74208444941084
// MI455X (gfx1250) — compile-verified, matrix operand order checked by reading, not run on hardware
//
#include <hip/hip_runtime.h>
#include <hip/hip_bf16.h>

typedef __attribute__((ext_vector_type(16))) _Float16 v16h;
typedef __attribute__((ext_vector_type(8)))  _Float16 v8h;
typedef __attribute__((ext_vector_type(4)))  _Float16 v4h;
typedef __attribute__((ext_vector_type(8)))  float    v8f;
typedef __attribute__((ext_vector_type(4)))  float    v4f;

// CDNA5 async global->LDS copy (raw bytes, ASYNCcnt-tracked).
// VDST = VGPR holding LDS byte address, VADDR = 64-bit global address.
#define ASYNC_LOAD_B128(ldsaddr, gptr)                                         \
  asm volatile("global_load_async_to_lds_b128 %0, %1, off"                     \
               :: "v"(ldsaddr), "v"(gptr) : "memory")
#define WAIT_ASYNC0() asm volatile("s_wait_asynccnt 0" ::: "memory")

__device__ __forceinline__ v16h cat8(v8h lo, v8h hi) {
  return __builtin_shufflevector(lo, hi, 0,1,2,3,4,5,6,7,8,9,10,11,12,13,14,15);
}

__device__ __forceinline__ v8f wmma_f16(v16h a, v16h b, v8f c) {
  // D(16x16 f32) = A(16x32 f16) * B(32x16 f16) + C
  return __builtin_amdgcn_wmma_f32_16x16x32_f16(false, a, false, b, (short)0, c,
                                                false, false);
}

__device__ __forceinline__ v8f zero8() {
  v8f z;
#pragma unroll
  for (int i = 0; i < 8; ++i) z[i] = 0.0f;
  return z;
}

// ---------------------------------------------------------------------------
// One-shot f32 -> f16 conversion (vectorized); frees GEMM inner loops from
// conversion VALU and halves repeated A/B global traffic.
// ---------------------------------------------------------------------------
__global__ __launch_bounds__(256, 1) void cvt_f32_f16_kernel(
    const float* __restrict__ src, _Float16* __restrict__ dst, int n4)
{
  int i = blockIdx.x * 256 + threadIdx.x;
  if (i < n4) {
    v4f d = *(const v4f*)(src + (size_t)i * 4);
    v4h h;
    h[0] = (_Float16)d[0]; h[1] = (_Float16)d[1];
    h[2] = (_Float16)d[2]; h[3] = (_Float16)d[3];
    *(v4h*)(dst + (size_t)i * 4) = h;
  }
}

// ---------------------------------------------------------------------------
// Tiled WMMA GEMM (all-f16 operands): C[M,N] = alpha * A[M,K] * B[N,K]^T
// WG = 256 threads (8 waves, 4M x 2N). Tile BM=128, BN=128, BK=32.
// Double-buffered async global->LDS staging; 8 WMMAs per wave per k-step.
// ---------------------------------------------------------------------------
template<bool OUT_F16, bool STORE_T>
__global__ __launch_bounds__(256, 1) void gemm_wmma_kernel(
    const _Float16* __restrict__ A, const _Float16* __restrict__ B,
    void* __restrict__ Cv, int M, int N, int K, float alpha)
{
  constexpr int BM = 128, BN = 128, BK = 32, PITCH = 40;  // 80B row pitch
  __shared__ alignas(16) _Float16 As[2][BM][PITCH];
  __shared__ alignas(16) _Float16 Bs[2][BN][PITCH];
  constexpr unsigned ABUF = BM * PITCH * 2;               // bytes per buffer
  constexpr unsigned BBUF = BN * PITCH * 2;

  const int tid  = threadIdx.x;
  const int wid  = tid >> 5;
  const int lane = tid & 31;
  const int lrow = lane & 15;
  const int kb   = (lane < 16) ? 0 : 8;
  const int mofs = (lane < 16) ? 0 : 8;
  const int wm = wid & 3, wn = wid >> 2;
  const int rbase = wm * 32;
  const int nbase = wn * 64;

  const int m0 = blockIdx.y * BM;
  const int n0 = blockIdx.x * BN;

  const unsigned asb = (unsigned)(uintptr_t)&As[0][0][0];
  const unsigned bsb = (unsigned)(uintptr_t)&Bs[0][0][0];

  // Stage one 128x32 f16 A tile + 128x32 B tile: 4 async b128 per thread.
  auto stage = [&](int k0, int buf) {
#pragma unroll
    for (int i = 0; i < 2; ++i) {
      int c = tid + i * 256;                 // 0..511 (4 x 16B chunks per row)
      int r = c >> 2, cc = c & 3;
      ASYNC_LOAD_B128(asb + (unsigned)buf * ABUF + (unsigned)(r * (PITCH * 2) + cc * 16),
                      A + (size_t)(m0 + r) * K + k0 + cc * 8);
    }
#pragma unroll
    for (int i = 0; i < 2; ++i) {
      int c = tid + i * 256;
      int r = c >> 2, cc = c & 3;
      ASYNC_LOAD_B128(bsb + (unsigned)buf * BBUF + (unsigned)(r * (PITCH * 2) + cc * 16),
                      B + (size_t)(n0 + r) * K + k0 + cc * 8);
    }
  };

  v8f acc[2][4];
#pragma unroll
  for (int mt = 0; mt < 2; ++mt)
#pragma unroll
    for (int t = 0; t < 4; ++t) acc[mt][t] = zero8();

  const int nk = K / BK;
  stage(0, 0);
  for (int kt = 0; kt < nk; ++kt) {
    WAIT_ASYNC0();                 // our stage kt copies done
    __syncthreads();               // everyone's done; prev compute finished
    if (kt + 1 < nk) stage((kt + 1) * BK, (kt + 1) & 1);  // overlaps compute
    const int buf = kt & 1;

    v16h af[2];
#pragma unroll
    for (int mt = 0; mt < 2; ++mt) {
      v8h lo = *(const v8h*)&As[buf][rbase + mt * 16 + lrow][kb];
      v8h hi = *(const v8h*)&As[buf][rbase + mt * 16 + lrow][kb + 16];
      af[mt] = cat8(lo, hi);
    }
#pragma unroll
    for (int t = 0; t < 4; ++t) {
      v8h blo = *(const v8h*)&Bs[buf][nbase + t * 16 + lrow][kb];
      v8h bhi = *(const v8h*)&Bs[buf][nbase + t * 16 + lrow][kb + 16];
      v16h bf = cat8(blo, bhi);
#pragma unroll
      for (int mt = 0; mt < 2; ++mt)
        acc[mt][t] = wmma_f16(af[mt], bf, acc[mt][t]);
    }
  }

  // ---- epilogue ----
#pragma unroll
  for (int mt = 0; mt < 2; ++mt) {
#pragma unroll
    for (int t = 0; t < 4; ++t) {
#pragma unroll
      for (int v = 0; v < 8; ++v) {
        int mm = m0 + rbase + mt * 16 + v + mofs;
        int nn = n0 + nbase + t * 16 + lrow;
        float val = acc[mt][t][v] * alpha;
        if (STORE_T) {
          ((_Float16*)Cv)[(size_t)nn * M + mm] = (_Float16)val;
        } else if (OUT_F16) {
          ((_Float16*)Cv)[(size_t)mm * N + nn] = (_Float16)val;
        } else {
          ((float*)Cv)[(size_t)mm * N + nn] = val;
        }
      }
    }
  }
}

// ---------------------------------------------------------------------------
// Causal flash attention v3: 4 waves per (head, 128-row q tile); wave w owns
// rows r0 = q0+32w .. r0+31 (2 m-frags). 32-row KV blocks staged once per WG
// via double-buffered async global->LDS copies, shared by all 4 waves.
// Per KV step per wave: 8 WMMAs (S=QK^T) + 8 WMMAs (O+=PV).
//   Q,Km : [S][1024] f16 (Q pre-scaled by 1/8), cols = h*64+d
//   Vt   : [1024][S] f16 (transposed) -> contiguous-K B fragments
//   Oa   : [S][1024] f16
// ---------------------------------------------------------------------------
__global__ __launch_bounds__(128, 1) void flash_attn_kernel(
    const _Float16* __restrict__ Q, const _Float16* __restrict__ Km,
    const _Float16* __restrict__ Vt, _Float16* __restrict__ Oa, int S)
{
  __shared__ alignas(16) _Float16 Ks[2][32][72];   // [kv][qk-dim 64], 144B pitch
  __shared__ alignas(16) _Float16 Vs[2][64][40];   // [vo-dim][kv 32], 80B pitch
  __shared__ alignas(16) _Float16 Ps[4][32][32];   // per-wave P bounce
  constexpr unsigned KBUF = 32 * 72 * 2;
  constexpr unsigned VBUF = 64 * 40 * 2;

  const int tid  = threadIdx.x;
  const int wid  = tid >> 5;
  const int lane = tid & 31;
  const int lrow = lane & 15;
  const int kb   = (lane < 16) ? 0 : 8;
  const int mofs = (lane < 16) ? 0 : 8;
  const int h    = blockIdx.y;
  const int q0   = blockIdx.x * 128;
  const int hcol = h * 64;
  const int r0   = q0 + wid * 32;                  // this wave's first q row

  const unsigned ksb = (unsigned)(uintptr_t)&Ks[0][0][0];
  const unsigned vsb = (unsigned)(uintptr_t)&Vs[0][0][0];

  // Stage K block (32x64) + V block (64x32): 4 async b128 per thread.
  auto stage = [&](int kvb, int buf) {
#pragma unroll
    for (int i = 0; i < 2; ++i) {
      int c = tid + i * 128;                       // 0..255, 8 chunks per K row
      int r = c >> 3, cc = c & 7;
      ASYNC_LOAD_B128(ksb + (unsigned)buf * KBUF + (unsigned)(r * 144 + cc * 16),
                      Km + (size_t)(kvb + r) * 1024 + hcol + cc * 8);
    }
#pragma unroll
    for (int i = 0; i < 2; ++i) {
      int c = tid + i * 128;                       // 4 chunks per V row
      int r = c >> 2, cc = c & 3;
      ASYNC_LOAD_B128(vsb + (unsigned)buf * VBUF + (unsigned)(r * 80 + cc * 16),
                      Vt + (size_t)(hcol + r) * S + kvb + cc * 8);
    }
  };

  // Q fragments: 2 m-tiles x 2 k-steps
  v16h qa[2][2];
#pragma unroll
  for (int mt = 0; mt < 2; ++mt) {
    const _Float16* qrow = Q + (size_t)(r0 + mt * 16 + lrow) * 1024 + hcol;
#pragma unroll
    for (int ks = 0; ks < 2; ++ks) {
      v8h lo = *(const v8h*)(qrow + ks * 32 + kb);
      v8h hi = *(const v8h*)(qrow + ks * 32 + kb + 16);
      qa[mt][ks] = cat8(lo, hi);
    }
  }

  float m_i[2][8], l_i[2][8];
  v8f o[2][4];
#pragma unroll
  for (int mt = 0; mt < 2; ++mt) {
#pragma unroll
    for (int v = 0; v < 8; ++v) { m_i[mt][v] = -1e30f; l_i[mt][v] = 0.0f; }
#pragma unroll
    for (int t = 0; t < 4; ++t) o[mt][t] = zero8();
  }

  const int jmax = (q0 + 127) >> 5;                // workgroup-wide KV bound
  stage(0, 0);
  for (int j = 0; j <= jmax; ++j) {
    WAIT_ASYNC0();                                 // stage j complete (ours)
    __syncthreads();                               // all waves' stage j done
    if (j < jmax) stage((j + 1) << 5, (j + 1) & 1);// overlaps compute below
    const int kvb = j << 5;
    const int buf = j & 1;

    if (kvb <= r0) {                               // wave-uniform causal guard
      // ---- S(32x32) = Q*K^T : 8 WMMAs ----
      v8f s[2][2];
#pragma unroll
      for (int mt = 0; mt < 2; ++mt)
#pragma unroll
        for (int nt = 0; nt < 2; ++nt) s[mt][nt] = zero8();
#pragma unroll
      for (int ks = 0; ks < 2; ++ks) {
#pragma unroll
        for (int nt = 0; nt < 2; ++nt) {
          v8h lo = *(const v8h*)&Ks[buf][nt * 16 + lrow][ks * 32 + kb];
          v8h hi = *(const v8h*)&Ks[buf][nt * 16 + lrow][ks * 32 + kb + 16];
          v16h bf = cat8(lo, hi);
#pragma unroll
          for (int mt = 0; mt < 2; ++mt)
            s[mt][nt] = wmma_f16(qa[mt][ks], bf, s[mt][nt]);
        }
      }

      // ---- causal mask (diagonal block only) ----
      if (kvb == r0) {
#pragma unroll
        for (int mt = 0; mt < 2; ++mt) {
#pragma unroll
          for (int v = 0; v < 8; ++v) {
            int row = r0 + mt * 16 + v + mofs;
            int c0 = kvb + lrow;
            if (c0 > row)      s[mt][0][v] = -1e30f;
            if (c0 + 16 > row) s[mt][1][v] = -1e30f;
          }
        }
      }

      // ---- online softmax per m-tile ----
#pragma unroll
      for (int mt = 0; mt < 2; ++mt) {
#pragma unroll
        for (int v = 0; v < 8; ++v) {
          float mv = fmaxf(s[mt][0][v], s[mt][1][v]);
          mv = fmaxf(mv, __shfl_xor(mv, 1));
          mv = fmaxf(mv, __shfl_xor(mv, 2));
          mv = fmaxf(mv, __shfl_xor(mv, 4));
          mv = fmaxf(mv, __shfl_xor(mv, 8));
          float mnew = fmaxf(m_i[mt][v], mv);
          float corr = __expf(m_i[mt][v] - mnew);
          m_i[mt][v] = mnew;
          float p0 = __expf(s[mt][0][v] - mnew);
          float p1 = __expf(s[mt][1][v] - mnew);
          float ps = p0 + p1;
          ps += __shfl_xor(ps, 1);
          ps += __shfl_xor(ps, 2);
          ps += __shfl_xor(ps, 4);
          ps += __shfl_xor(ps, 8);
          l_i[mt][v] = l_i[mt][v] * corr + ps;
#pragma unroll
          for (int t = 0; t < 4; ++t) o[mt][t][v] *= corr;
          Ps[wid][mt * 16 + v + mofs][lrow]      = (_Float16)p0;
          Ps[wid][mt * 16 + v + mofs][lrow + 16] = (_Float16)p1;
        }
      }

      // ---- O(32x64) += P(32x32) * V(32x64) : 8 WMMAs ----
      // (same-wave LDS ops are in-order: Ps readback needs no barrier)
      v16h pa[2];
#pragma unroll
      for (int mt = 0; mt < 2; ++mt) {
        v8h plo = *(const v8h*)&Ps[wid][mt * 16 + lrow][kb];
        v8h phi = *(const v8h*)&Ps[wid][mt * 16 + lrow][kb + 16];
        pa[mt] = cat8(plo, phi);
      }
#pragma unroll
      for (int t = 0; t < 4; ++t) {
        v8h lo = *(const v8h*)&Vs[buf][t * 16 + lrow][kb];
        v8h hi = *(const v8h*)&Vs[buf][t * 16 + lrow][kb + 16];
        v16h vb = cat8(lo, hi);
#pragma unroll
        for (int mt = 0; mt < 2; ++mt)
          o[mt][t] = wmma_f16(pa[mt], vb, o[mt][t]);
      }
    }
  }

  // ---- normalize + store attn ----
#pragma unroll
  for (int mt = 0; mt < 2; ++mt) {
#pragma unroll
    for (int v = 0; v < 8; ++v) {
      float inv = 1.0f / l_i[mt][v];
      int row = r0 + mt * 16 + v + mofs;
#pragma unroll
      for (int t = 0; t < 4; ++t) {
        Oa[(size_t)row * 1024 + hcol + t * 16 + lrow] = (_Float16)(o[mt][t][v] * inv);
      }
    }
  }
}

extern "C" void kernel_launch(void* const* d_in, const int* in_sizes, int n_in,
                              void* d_out, int out_size, void* d_ws, size_t ws_size,
                              hipStream_t stream) {
  constexpr int S = 4096, D = 1024;      // NUM_HEADS*QK_SIZE == D
  const float* query = (const float*)d_in[0];
  const float* key   = (const float*)d_in[1];
  const float* value = (const float*)d_in[2];
  const float* Wq    = (const float*)d_in[3];
  const float* Wk    = (const float*)d_in[4];
  const float* Wv    = (const float*)d_in[5];
  const float* Wo    = (const float*)d_in[6];
  float* out = (float*)d_out;

  const size_t SD = (size_t)S * D, DD = (size_t)D * D;
  _Float16* p = (_Float16*)d_ws;
  _Float16* qf  = p; p += SD;            // f16 copies of inputs
  _Float16* kf  = p; p += SD;
  _Float16* vf  = p; p += SD;
  _Float16* wqf = p; p += DD;
  _Float16* wkf = p; p += DD;
  _Float16* wvf = p; p += DD;
  _Float16* wof = p; p += DD;
  _Float16* Qb  = p; p += SD;            // [S][D] f16 (pre-scaled)
  _Float16* Kb  = p; p += SD;            // [S][D] f16
  _Float16* Vt  = p; p += SD;            // [D][S] f16 (transposed)
  _Float16* Ab  = p; p += SD;            // [S][D] f16 attn output

  // f32 -> f16 conversion passes (bandwidth-bound)
  const int gSD = (int)(SD / 4) / 256, gDD = (int)(DD / 4) / 256;
  cvt_f32_f16_kernel<<<gSD, 256, 0, stream>>>(query, qf,  (int)(SD / 4));
  cvt_f32_f16_kernel<<<gSD, 256, 0, stream>>>(key,   kf,  (int)(SD / 4));
  cvt_f32_f16_kernel<<<gSD, 256, 0, stream>>>(value, vf,  (int)(SD / 4));
  cvt_f32_f16_kernel<<<gDD, 256, 0, stream>>>(Wq,    wqf, (int)(DD / 4));
  cvt_f32_f16_kernel<<<gDD, 256, 0, stream>>>(Wk,    wkf, (int)(DD / 4));
  cvt_f32_f16_kernel<<<gDD, 256, 0, stream>>>(Wv,    wvf, (int)(DD / 4));
  cvt_f32_f16_kernel<<<gDD, 256, 0, stream>>>(Wo,    wof, (int)(DD / 4));

  dim3 gg(D / 128, S / 128);             // GEMM grid (8, 32)

  // Projections (softmax scale 1/sqrt(64) folded into Q)
  gemm_wmma_kernel<true, false><<<gg, 256, 0, stream>>>(qf, wqf, Qb, S, D, D, 0.125f);
  gemm_wmma_kernel<true, false><<<gg, 256, 0, stream>>>(kf, wkf, Kb, S, D, D, 1.0f);
  gemm_wmma_kernel<true, true ><<<gg, 256, 0, stream>>>(vf, wvf, Vt, S, D, D, 1.0f);

  // Causal flash attention: 32 q-tiles x 16 heads, 4 waves each
  flash_attn_kernel<<<dim3(S / 128, 16), 128, 0, stream>>>(Qb, Kb, Vt, Ab, S);

  // Output projection -> f32 d_out
  gemm_wmma_kernel<false, false><<<gg, 256, 0, stream>>>(Ab, wof, out, S, D, D, 1.0f);
}
